// NeuralSDE_38680475468496
// MI455X (gfx1250) — compile-verified, matrix operand order checked by reading, not run on hardware
//
#include <hip/hip_runtime.h>
#include <cmath>
#include <cstddef>
#include <cstdint>

typedef __attribute__((ext_vector_type(16))) _Float16 v16h;
typedef __attribute__((ext_vector_type(8)))  float    v8f;

#define M_PI_F 3.14159265358979323846f

constexpr int BN  = 4096;
constexpr int TN  = 40;
constexpr int HN  = 128;
constexpr int GHN = 384;         // 3*H gate width
constexpr int BTN = BN * TN;
constexpr int NGRID = 100;

// LDS layout for the persistent GRU kernel (dynamic shared memory):
//   [0, 96KB)        : Whh^T B-fragments, f16  [nt(24)][kb(4)][lane(32)][e(16)]
//   [96KB, 112KB)    : hidden-state A-fragments, f16 [row_tile(4)][kb(4)][lane(32)][e(16)]
constexpr int WLDS_HALFS  = 24 * 4 * 32 * 16;      // 49152 halves = 96KB
constexpr int AFRAG_HALFS = 4 * 4 * 32 * 16;       // 8192 halves = 16KB
constexpr size_t GRU_LDS_BYTES = (size_t)WLDS_HALFS * 2 + (size_t)AFRAG_HALFS * 2;

// ---------------------------------------------------------------------------
// WMMA fragment layout (16-bit A/B layout from CDNA5 ISA 7.12.2):
//   A tile (16 rows M x 32 K):  lane = M + 16*((K>>3)&1),  elem = (K&7) + 8*(K>>4)
//   B tile (32 K x 16 N):       lane = N + 16*((K>>3)&1),  elem = (K&7) + 8*(K>>4)
// Fragments stored as [tile][lane(32)][elem(16)] contiguous halves (32B/lane).
// ---------------------------------------------------------------------------

// Swizzle a [384,128] f32 weight (gate-major rows g, K cols) into the
// B-fragment f16 image: [nt(24)][kb(4)][lane(32)][e(16)].
__global__ void swizzle_w_kernel(const float* __restrict__ src,
                                 _Float16* __restrict__ dst) {
  int i = blockIdx.x * blockDim.x + threadIdx.x;
  if (i >= GHN * HN) return;
  int g = i >> 7, k = i & 127;
  int nt = g >> 4, n = g & 15;
  int kb = k >> 5, kk = k & 31;
  int fl = n + ((kk >> 3) & 1) * 16;
  int fe = (kk & 7) + ((kk >> 4) << 3);
  dst[(size_t)((nt * 4 + kb) * 32 + fl) * 16 + fe] = (_Float16)src[i];
}

// Layer-0 input projection, F=1: xp[b,t,g] = x[b,t]*Wih0[g] + bih0[g]
__global__ void xp0_kernel(const float* __restrict__ x,
                           const float* __restrict__ Wih0,
                           const float* __restrict__ bih0,
                           float* __restrict__ xp) {
  size_t i = (size_t)blockIdx.x * blockDim.x + threadIdx.x;
  if (i >= (size_t)BTN * GHN) return;
  size_t bt = i / GHN;
  int g = (int)(i % GHN);
  xp[i] = x[bt] * Wih0[g] + bih0[g];
}

// ---------------------------------------------------------------------------
// Persistent GRU layer kernel.  One WG = 64 batch rows, 16 waves (512 thr).
// Recurrent weights live in LDS (96KB, staged once, reused 40 steps).
// Hidden state lives in LDS as WMMA A-fragments (f16).  Per step:
//   gh = h @ Whh^T  via V_WMMA_F32_16X16X32_F16 (K=128 -> 4 k-steps)
// Wave (wr,wc): rows [wr*16,+16), h-cols [wc*32,+32) for all 3 gates
// => the gate fusion needs no inter-wave communication.
// ---------------------------------------------------------------------------
__global__ __launch_bounds__(512) void gru_layer_kernel(
    const float* __restrict__ xp,      // [B,T,384] gate inputs (incl. bih)
    const _Float16* __restrict__ Wsw,  // swizzled Whh^T B-fragments (global)
    const float* __restrict__ bhh,     // [384]
    float* __restrict__ s_out,         // [B,T,128] per-step outputs
    float* __restrict__ enc_h) {       // final h -> enc[b*384 + col], col 0..127
  extern __shared__ __align__(32) char smem[];
  _Float16* wlds = (_Float16*)smem;                          // 96KB weights
  v16h* afrag = (v16h*)(smem + (size_t)WLDS_HALFS * 2);      // 16KB h-fragments
  _Float16* afh = (_Float16*)afrag;

  const int tid  = threadIdx.x;
  const int lane = tid & 31;
  const int w    = tid >> 5;     // 0..15
  const int wr   = w >> 2;       // row-tile 0..3
  const int wc   = w & 3;        // h-col group 0..3
  const int row0 = blockIdx.x * 64;
  const int ncol  = lane & 15;
  const int mhalf = (lane >> 4) * 8;

  // stage weights global -> LDS (128-bit copies) and zero the h fragments
  {
    const uint4* src = (const uint4*)Wsw;
    uint4* dst = (uint4*)wlds;
    for (int i = tid; i < WLDS_HALFS / 8; i += 512) dst[i] = src[i];
  }
  for (int i = tid; i < AFRAG_HALFS; i += 512) afh[i] = (_Float16)0.0f;

  float bhh_v[2][3];
  for (int s = 0; s < 2; ++s)
    for (int g = 0; g < 3; ++g)
      bhh_v[s][g] = bhh[g * HN + wc * 32 + s * 16 + ncol];

  float hprev[2][8];
  for (int s = 0; s < 2; ++s)
    for (int j = 0; j < 8; ++j) hprev[s][j] = 0.0f;

  __syncthreads();

  unsigned opq = 0;  // opaque 0: defeats LICM of the LDS weight loads
  for (int t = 0; t < TN; ++t) {
    asm volatile("" : "+v"(opq));
    const _Float16* wl = wlds + opq;

    if (t + 1 < TN) {  // warm L2/WGP$ for next step's gate inputs
      const float* pf = xp + ((size_t)(row0 + wr * 16 + mhalf) * TN + (t + 1)) * GHN
                           + wc * 32 + ncol;
      __builtin_prefetch(pf, 0, 1);
    }
    v8f zf = {};
    v8f acc[2][3];
    for (int s = 0; s < 2; ++s)
      for (int g = 0; g < 3; ++g) acc[s][g] = zf;
#pragma unroll
    for (int kb = 0; kb < 4; ++kb) {
      v16h a = afrag[(wr * 4 + kb) * 32 + lane];
#pragma unroll
      for (int s = 0; s < 2; ++s)
#pragma unroll
        for (int g = 0; g < 3; ++g) {
          int nt = g * 8 + wc * 2 + s;
          v16h bf = *(const v16h*)(wl + (size_t)((nt * 4 + kb) * 32 + lane) * 16);
          acc[s][g] = __builtin_amdgcn_wmma_f32_16x16x32_f16(
              false, a, false, bf, (short)0, acc[s][g], false, false);
        }
    }
    __syncthreads();  // all A-fragment reads done before overwrite
#pragma unroll
    for (int s = 0; s < 2; ++s) {
      const int hcol = wc * 32 + s * 16 + ncol;
      const int kb2  = hcol >> 5;              // == wc
      const int kk   = hcol & 31;
      const int fe   = (kk & 7) + ((kk >> 4) << 3);
      const int flk  = ((kk >> 3) & 1) * 16;
#pragma unroll
      for (int j = 0; j < 8; ++j) {
        int m = wr * 16 + mhalf + j;
        int b = row0 + m;
        const float* xr = xp + ((size_t)b * TN + t) * GHN;
        float hr = acc[s][0][j] + bhh_v[s][0];
        float hz = acc[s][1][j] + bhh_v[s][1];
        float hn = acc[s][2][j] + bhh_v[s][2];
        float r = 1.0f / (1.0f + expf(-(xr[hcol] + hr)));
        float z = 1.0f / (1.0f + expf(-(xr[HN + hcol] + hz)));
        float n = tanhf(xr[2 * HN + hcol] + r * hn);
        float hnew = (1.0f - z) * n + z * hprev[s][j];
        hprev[s][j] = hnew;
        s_out[((size_t)b * TN + t) * HN + hcol] = hnew;
        if (t == TN - 1) enc_h[(size_t)b * GHN + hcol] = hnew;
        int fl = (m & 15) + flk;
        afh[(size_t)((wr * 4 + kb2) * 32 + fl) * 16 + fe] = (_Float16)hnew;
      }
    }
    __syncthreads();  // new fragments visible before next step's WMMA
  }
}

// ---------------------------------------------------------------------------
// Tiled GEMM: out[M,384] = A[M,128] @ Wsw + bias   (layer-1 input projection)
// ---------------------------------------------------------------------------
__global__ __launch_bounds__(512) void gemm_xp_kernel(
    const float* __restrict__ A, const _Float16* __restrict__ Wsw,
    const float* __restrict__ bias, float* __restrict__ out, int Mtot) {
  __shared__ v16h afrag[4 * 4 * 32];
  _Float16* afh = (_Float16*)afrag;
  const int tid = threadIdx.x, lane = tid & 31, w = tid >> 5;
  const int wr = w >> 2, wc = w & 3;
  const int row0 = blockIdx.x * 64;
  const int ncol = lane & 15, mhalf = (lane >> 4) * 8;

  for (int i = tid; i < 64 * 128; i += 512) {
    int m = i >> 7, k = i & 127;
    float v = (row0 + m < Mtot) ? A[(size_t)(row0 + m) * HN + k] : 0.0f;
    int fl = (m & 15) + ((k >> 3) & 1) * 16;
    int fe = (k & 7) + ((k >> 4) << 3);
    afh[(size_t)(((m >> 4) * 4 + (k >> 5)) * 32 + fl) * 16 + fe] = (_Float16)v;
  }
  __syncthreads();

  v8f zf = {};
  v8f acc[2][3];
  for (int s = 0; s < 2; ++s)
    for (int g = 0; g < 3; ++g) acc[s][g] = zf;
#pragma unroll
  for (int kb = 0; kb < 4; ++kb) {
    v16h a = afrag[(wr * 4 + kb) * 32 + lane];
#pragma unroll
    for (int s = 0; s < 2; ++s)
#pragma unroll
      for (int g = 0; g < 3; ++g) {
        int nt = g * 8 + wc * 2 + s;
        v16h bf = *(const v16h*)(Wsw + (size_t)((nt * 4 + kb) * 32 + lane) * 16);
        acc[s][g] = __builtin_amdgcn_wmma_f32_16x16x32_f16(
            false, a, false, bf, (short)0, acc[s][g], false, false);
      }
  }
#pragma unroll
  for (int s = 0; s < 2; ++s)
#pragma unroll
    for (int j = 0; j < 8; ++j) {
      int row = row0 + wr * 16 + mhalf + j;
      if (row >= Mtot) continue;
      for (int g = 0; g < 3; ++g) {
        int col = g * HN + wc * 32 + s * 16 + ncol;
        out[(size_t)row * GHN + col] = acc[s][g][j] + bias[col];
      }
    }
}

// mean over T of top-layer sequence -> enc[:,256:384] (enc pre-offset)
__global__ void seqmean_kernel(const float* __restrict__ s1,
                               float* __restrict__ enc_mean) {
  int i = blockIdx.x * blockDim.x + threadIdx.x;
  if (i >= BN * HN) return;
  int b = i >> 7, h = i & 127;
  float acc = 0.0f;
  for (int t = 0; t < TN; ++t) acc += s1[((size_t)b * TN + t) * HN + h];
  enc_mean[(size_t)b * GHN + h] = acc * (1.0f / TN);
}

// head: per-row dot(enc[row,:384], Wh) + bh; clamp; mode 1 -> softplus+1e-4
__global__ void head_kernel(const float* __restrict__ enc,
                            const float* __restrict__ Wh,
                            const float* __restrict__ bh,
                            float* __restrict__ out, int mode) {
  int lane = threadIdx.x & 31;
  int w = threadIdx.x >> 5;
  int row = blockIdx.x * 8 + w;
  if (row >= BN) return;
  float s = 0.0f;
  for (int i = lane; i < GHN; i += 32) s += enc[(size_t)row * GHN + i] * Wh[i];
  for (int off = 16; off > 0; off >>= 1) s += __shfl_xor(s, off, 32);
  if (lane == 0) {
    float v = fminf(fmaxf(s + bh[0], -20.0f), 20.0f);
    out[row] = mode ? (log1pf(expf(v)) + 1e-4f) : v;
  }
}

// ----- Student-t tail: NLL + CDF (regularized incomplete beta) -------------
__device__ float betacf_dev(float a, float b, float x) {
  const float FPMIN = 1e-30f;
  float qab = a + b, qap = a + 1.0f, qam = a - 1.0f;
  float c = 1.0f, d = 1.0f - qab * x / qap;
  if (fabsf(d) < FPMIN) d = FPMIN;
  d = 1.0f / d;
  float h = d;
  for (int m = 1; m <= 100; ++m) {
    float m2 = 2.0f * m;
    float aa = m * (b - m) * x / ((qam + m2) * (a + m2));
    d = 1.0f + aa * d; if (fabsf(d) < FPMIN) d = FPMIN;
    c = 1.0f + aa / c; if (fabsf(c) < FPMIN) c = FPMIN;
    d = 1.0f / d; h *= d * c;
    aa = -(a + m) * (qab + m) * x / ((a + m2) * (qap + m2));
    d = 1.0f + aa * d; if (fabsf(d) < FPMIN) d = FPMIN;
    c = 1.0f + aa / c; if (fabsf(c) < FPMIN) c = FPMIN;
    d = 1.0f / d;
    float del = d * c; h *= del;
    if (fabsf(del - 1.0f) < 3e-7f) break;
  }
  return h;
}
__device__ float betainc_reg(float a, float b, float x) {
  if (x <= 0.0f) return 0.0f;
  if (x >= 1.0f) return 1.0f;
  float lbeta = lgammaf(a) + lgammaf(b) - lgammaf(a + b);
  float front = expf(a * logf(x) + b * logf(1.0f - x) - lbeta);
  if (x < (a + 1.0f) / (a + b + 2.0f)) return front * betacf_dev(a, b, x) / a;
  return 1.0f - front * betacf_dev(b, a, 1.0f - x) / b;
}

// acc[0]=nll_sum acc[1]=cdf_sum acc[2]=cdf_sumsq acc[3]=penalty acc[4]=hbw
__global__ void stats_kernel(const float* __restrict__ dX,
                             const float* __restrict__ dTp,
                             const float* __restrict__ mu,
                             const float* __restrict__ sigma,
                             const float* __restrict__ lognu,
                             float* __restrict__ cdf, float* acc) {
  int b = blockIdx.x * blockDim.x + threadIdx.x;
  float nll = 0.0f, cs = 0.0f, cs2 = 0.0f;
  if (b < BN) {
    float df = fminf(fmaxf(expf(lognu[0]), 2.1f), 30.0f);
    float dt = dTp[0];
    float scale = sigma[b] * sqrtf(dt);
    float y = (dX[b] - mu[b] * dt) / scale;
    float lp = lgammaf(0.5f * (df + 1.0f)) - lgammaf(0.5f * df)
             - 0.5f * logf(df * M_PI_F) - logf(scale)
             - 0.5f * (df + 1.0f) * log1pf(y * y / df);
    nll = -lp;
    float xb = df / (df + y * y);
    float ib = betainc_reg(0.5f * df, 0.5f, xb);
    float c = (y <= 0.0f) ? 0.5f * ib : 1.0f - 0.5f * ib;
    c = fminf(fmaxf(c, 1e-6f), 1.0f - 1e-6f);
    cdf[b] = c;
    cs = c;
    cs2 = c * c;
  }
  for (int off = 16; off > 0; off >>= 1) {
    nll += __shfl_xor(nll, off, 32);
    cs  += __shfl_xor(cs,  off, 32);
    cs2 += __shfl_xor(cs2, off, 32);
  }
  if ((threadIdx.x & 31) == 0) {
    atomicAdd(&acc[0], nll);
    atomicAdd(&acc[1], cs);
    atomicAdd(&acc[2], cs2);
  }
}

__global__ void hbw_kernel(float* acc) {
  float mean = acc[1] / (float)BN;
  float var = fmaxf((acc[2] - (float)BN * mean * mean) / (float)(BN - 1), 0.0f);
  acc[4] = 1.06f * sqrtf(var) * powf((float)BN, -0.2f) + 1e-5f;
}

__global__ void kde_kernel(const float* __restrict__ cdf, float* acc) {
  __shared__ float red[256];
  float u = (float)blockIdx.x / (float)(NGRID - 1);
  float hbw = acc[4];
  float inv = 1.0f / hbw;
  float s = 0.0f;
  for (int b = threadIdx.x; b < BN; b += 256) {
    float d = (u - cdf[b]) * inv;
    s += expf(-0.5f * d * d);
  }
  red[threadIdx.x] = s;
  __syncthreads();
  for (int st = 128; st > 0; st >>= 1) {
    if (threadIdx.x < st) red[threadIdx.x] += red[threadIdx.x + st];
    __syncthreads();
  }
  if (threadIdx.x == 0) {
    float f_hat = red[0] / 2.50662827f / ((float)BN * hbw);
    float dv = f_hat - 1.0f;
    atomicAdd(&acc[3], dv * dv * (1.0f / NGRID));
  }
}

__global__ void init_acc_kernel(float* acc) {
  if (threadIdx.x < 8) acc[threadIdx.x] = 0.0f;
}

__global__ void final_kernel(const float* acc, float* out) {
  out[0] = acc[0] / (float)BN + 100.0f * acc[3];
}

// ---------------------------------------------------------------------------
extern "C" void kernel_launch(void* const* d_in, const int* in_sizes, int n_in,
                              void* d_out, int out_size, void* d_ws, size_t ws_size,
                              hipStream_t stream) {
  (void)in_sizes; (void)n_in; (void)out_size; (void)ws_size;
  const float* x    = (const float*)d_in[0];
  const float* dX   = (const float*)d_in[1];
  const float* dTp  = (const float*)d_in[2];
  const float* Wp[2][8];
  const float* Whd[2];
  const float* bhd[2];
  for (int e = 0; e < 2; ++e) {
    int base = 3 + e * 10;  // d_ params then s_ params
    for (int k = 0; k < 8; ++k) Wp[e][k] = (const float*)d_in[base + k];
    Whd[e] = (const float*)d_in[base + 8];
    bhd[e] = (const float*)d_in[base + 9];
  }
  const float* lognu = (const float*)d_in[23];
  float* out = (float*)d_out;

  // workspace carve (256B aligned)
  char* ws = (char*)d_ws;
  size_t o = 0;
  auto carve = [&](size_t bytes) -> void* {
    void* p = ws + o;
    o += (bytes + 255) & ~(size_t)255;
    return p;
  };
  _Float16* wsw[2][3];  // [enc][0=Whh0, 1=Wih1, 2=Whh1] swizzled f16
  for (int e = 0; e < 2; ++e)
    for (int m = 0; m < 3; ++m)
      wsw[e][m] = (_Float16*)carve((size_t)GHN * HN * sizeof(_Float16));
  float* xp  = (float*)carve((size_t)BTN * GHN * sizeof(float));
  float* sA  = (float*)carve((size_t)BTN * HN * sizeof(float));
  float* sB  = (float*)carve((size_t)BTN * HN * sizeof(float));
  float* enc[2];
  enc[0] = (float*)carve((size_t)BN * GHN * sizeof(float));
  enc[1] = (float*)carve((size_t)BN * GHN * sizeof(float));
  float* mu    = (float*)carve(BN * sizeof(float));
  float* sigma = (float*)carve(BN * sizeof(float));
  float* cdf   = (float*)carve(BN * sizeof(float));
  float* acc   = (float*)carve(64 * sizeof(float));

  const int SWZ_B = (GHN * HN + 255) / 256;
  for (int e = 0; e < 2; ++e) {
    swizzle_w_kernel<<<SWZ_B, 256, 0, stream>>>(Wp[e][1], wsw[e][0]);  // Whh0
    swizzle_w_kernel<<<SWZ_B, 256, 0, stream>>>(Wp[e][4], wsw[e][1]);  // Wih1
    swizzle_w_kernel<<<SWZ_B, 256, 0, stream>>>(Wp[e][5], wsw[e][2]);  // Whh1
  }

  const size_t xpN = (size_t)BTN * GHN;
  const int XP_B = (int)((xpN + 255) / 256);
  float* heads[2] = {mu, sigma};
  for (int e = 0; e < 2; ++e) {
    // Wp[e]: 0=Wih0 1=Whh0 2=bih0 3=bhh0 4=Wih1 5=Whh1 6=bih1 7=bhh1
    xp0_kernel<<<XP_B, 256, 0, stream>>>(x, Wp[e][0], Wp[e][2], xp);
    gru_layer_kernel<<<BN / 64, 512, GRU_LDS_BYTES, stream>>>(
        xp, wsw[e][0], Wp[e][3], sA, enc[e]);
    gemm_xp_kernel<<<BTN / 64, 512, 0, stream>>>(sA, wsw[e][1], Wp[e][6], xp, BTN);
    gru_layer_kernel<<<BN / 64, 512, GRU_LDS_BYTES, stream>>>(
        xp, wsw[e][2], Wp[e][7], sB, enc[e] + HN);
    seqmean_kernel<<<(BN * HN + 255) / 256, 256, 0, stream>>>(sB, enc[e] + 2 * HN);
    head_kernel<<<BN / 8, 256, 0, stream>>>(enc[e], Whd[e], bhd[e], heads[e], e);
  }
  init_acc_kernel<<<1, 32, 0, stream>>>(acc);
  stats_kernel<<<BN / 256, 256, 0, stream>>>(dX, dTp, mu, sigma, lognu, cdf, acc);
  hbw_kernel<<<1, 1, 0, stream>>>(acc);
  kde_kernel<<<NGRID, 256, 0, stream>>>(cdf, acc);
  final_kernel<<<1, 1, 0, stream>>>(acc, out);
}
